// PointNetRefinePoint_51891794870661
// MI455X (gfx1250) — matrix operand order repaired and checked by reading, not compiled, not run on hardware
//
#include <hip/hip_runtime.h>
#include <hip/hip_bf16.h>

// ---------------------------------------------------------------------------
// PointNet++ refine net for MI455X (gfx1250, wave32).
// MLP layers run as v_wmma_f32_16x16x32_f16 GEMMs (f16 in, f32 accumulate),
// one 16x64 output strip per wave with a 2-stage software pipeline so loads
// overlap matrix ops.  Geometry (FPS / ball query / 3-NN) are latency-bound
// scan kernels; everything stays L2-resident (max buffer 67MB < 192MB L2).
// ---------------------------------------------------------------------------

typedef __attribute__((ext_vector_type(16))) _Float16 v16h;
typedef __attribute__((ext_vector_type(8)))  float    v8f;
typedef unsigned int u32;

static constexpr int BATCH = 4;
static constexpr int NPTS  = 8192;

// ------------------------------ geometry -----------------------------------

__global__ void transpose_in_kernel(const float* __restrict__ xyz,
                                    const float* __restrict__ nxyz,
                                    const float* __restrict__ pts,
                                    float* __restrict__ xyz0,
                                    float* __restrict__ nxyz0,
                                    float* __restrict__ feat0) {
  int i = blockIdx.x * blockDim.x + threadIdx.x;   // b*N + j
  if (i >= BATCH * NPTS) return;
  int b = i / NPTS, j = i % NPTS;
  for (int d = 0; d < 3; ++d) {
    xyz0 [i * 3 + d] = xyz [((size_t)b * 3 + d) * NPTS + j];
    nxyz0[i * 3 + d] = nxyz[((size_t)b * 3 + d) * NPTS + j];
  }
  for (int d = 0; d < 6; ++d)
    feat0[(size_t)i * 6 + d] = pts[((size_t)b * 6 + d) * NPTS + j];
}

// Farthest point sampling: one block per batch, sequential npoint iterations.
__global__ void fps_kernel(const float* __restrict__ xyz, int n, int npoint,
                           int* __restrict__ fidx, float* __restrict__ dist) {
  int b = blockIdx.x;
  const float* X = xyz + (size_t)b * n * 3;
  float* D = dist + (size_t)b * n;
  __shared__ float smax[256];
  __shared__ int   simx[256];
  __shared__ int   s_far;
  int t = threadIdx.x;
  for (int i = t; i < n; i += 256) D[i] = 1e10f;
  if (t == 0) s_far = 0;
  __syncthreads();
  for (int it = 0; it < npoint; ++it) {
    int far = s_far;
    if (t == 0) fidx[b * npoint + it] = far;
    float cx = X[far * 3 + 0], cy = X[far * 3 + 1], cz = X[far * 3 + 2];
    float bm = -1.f; int bi = 0;
    for (int i = t; i < n; i += 256) {
      float dx = X[i * 3 + 0] - cx, dy = X[i * 3 + 1] - cy, dz = X[i * 3 + 2] - cz;
      float d = fminf(D[i], dx * dx + dy * dy + dz * dz);
      D[i] = d;
      if (d > bm) { bm = d; bi = i; }
    }
    smax[t] = bm; simx[t] = bi;
    __syncthreads();
    for (int s = 128; s > 0; s >>= 1) {
      if (t < s) {
        if (smax[t + s] > smax[t] ||
            (smax[t + s] == smax[t] && simx[t + s] < simx[t])) {
          smax[t] = smax[t + s]; simx[t] = simx[t + s];
        }
      }
      __syncthreads();
    }
    if (t == 0) s_far = simx[0];
    __syncthreads();
  }
}

__global__ void gather_pts_kernel(const float* __restrict__ xyz,
                                  const float* __restrict__ nxyz,
                                  const int* __restrict__ fidx,
                                  int n, int npoint,
                                  float* __restrict__ nx, float* __restrict__ nn,
                                  int total) {
  int i = blockIdx.x * blockDim.x + threadIdx.x;
  if (i >= total) return;
  int b = i / npoint;
  int j = fidx[i];
  for (int d = 0; d < 3; ++d) {
    nx[i * 3 + d] = xyz [((size_t)b * n + j) * 3 + d];
    nn[i * 3 + d] = nxyz[((size_t)b * n + j) * 3 + d];
  }
}

// First `ns` in-radius neighbors in increasing index order (== reference's
// sorted-then-truncated semantics), padded with the first hit.
__global__ void ball_query_kernel(const float* __restrict__ xyz,
                                  const float* __restrict__ new_xyz,
                                  int n, int npoint, int ns, float r2,
                                  int* __restrict__ gidx, int total) {
  int idx = blockIdx.x * blockDim.x + threadIdx.x;   // b*npoint + s
  if (idx >= total) return;
  int b = idx / npoint;
  const float* X = xyz + (size_t)b * n * 3;
  float qx = new_xyz[idx * 3 + 0], qy = new_xyz[idx * 3 + 1], qz = new_xyz[idx * 3 + 2];
  int* out = gidx + (size_t)idx * ns;
  int cnt = 0, first = 0;
  for (int j = 0; j < n && cnt < ns; ++j) {
    float dx = X[j * 3 + 0] - qx, dy = X[j * 3 + 1] - qy, dz = X[j * 3 + 2] - qz;
    if (dx * dx + dy * dy + dz * dz <= r2) {
      if (cnt == 0) first = j;
      out[cnt++] = j;
    }
  }
  if (cnt == 0) out[cnt++] = 0;
  for (; cnt < ns; ++cnt) out[cnt] = first;
}

// Build SA grouped input rows: [g_xyz-center(3), g_nxyz(3), feat(Cf)] -> f16.
__global__ void build_sa_input_kernel(const float* __restrict__ xyz,
                                      const float* __restrict__ nxyz,
                                      const float* __restrict__ feat, int Cf,
                                      const float* __restrict__ new_xyz,
                                      const int* __restrict__ gidx,
                                      int n, int npoint, int ns, int Kpad,
                                      _Float16* __restrict__ X0, int M) {
  int row = blockIdx.x * blockDim.x + threadIdx.x;
  if (row >= M) return;
  int s = (row / ns) % npoint;
  int b = row / (ns * npoint);
  int j = gidx[row];
  const float* xp = xyz  + ((size_t)b * n + j) * 3;
  const float* np_ = nxyz + ((size_t)b * n + j) * 3;
  const float* q  = new_xyz + ((size_t)b * npoint + s) * 3;
  const float* f  = feat + ((size_t)b * n + j) * Cf;
  _Float16* o = X0 + (size_t)row * Kpad;
  o[0] = (_Float16)(xp[0] - q[0]);
  o[1] = (_Float16)(xp[1] - q[1]);
  o[2] = (_Float16)(xp[2] - q[2]);
  o[3] = (_Float16)np_[0];
  o[4] = (_Float16)np_[1];
  o[5] = (_Float16)np_[2];
  for (int c = 0; c < Cf; ++c) o[6 + c] = (_Float16)f[c];
  for (int c = 6 + Cf; c < Kpad; ++c) o[c] = (_Float16)0.f;
}

// 3-NN interpolation + FP concat row build:
// [xyz1(3), nxyz1(3), pts1(C1), interp_n(3), interp(C2)] -> f16.
__global__ void fp_build_kernel(const float* __restrict__ xyz1,
                                const float* __restrict__ nxyz1,
                                const float* __restrict__ pts1, int C1,
                                const float* __restrict__ xyz2,
                                const float* __restrict__ nxyz2,
                                const float* __restrict__ pts2, int C2,
                                int n1, int n2, int Kpad,
                                _Float16* __restrict__ X0, int total) {
  int idx = blockIdx.x * blockDim.x + threadIdx.x;   // b*n1 + i
  if (idx >= total) return;
  int b = idx / n1;
  float qx = xyz1[idx * 3 + 0], qy = xyz1[idx * 3 + 1], qz = xyz1[idx * 3 + 2];
  const float* X2 = xyz2 + (size_t)b * n2 * 3;
  float bd0 = 1e30f, bd1 = 1e30f, bd2 = 1e30f;
  int   bi0 = 0,     bi1 = 0,     bi2 = 0;
  for (int j = 0; j < n2; ++j) {
    float dx = X2[j * 3 + 0] - qx, dy = X2[j * 3 + 1] - qy, dz = X2[j * 3 + 2] - qz;
    float d = dx * dx + dy * dy + dz * dz;
    if (d < bd0)      { bd2 = bd1; bi2 = bi1; bd1 = bd0; bi1 = bi0; bd0 = d; bi0 = j; }
    else if (d < bd1) { bd2 = bd1; bi2 = bi1; bd1 = d; bi1 = j; }
    else if (d < bd2) { bd2 = d; bi2 = j; }
  }
  float w0 = 1.f / (bd0 + 1e-8f), w1 = 1.f / (bd1 + 1e-8f), w2 = 1.f / (bd2 + 1e-8f);
  float ws = w0 + w1 + w2;
  w0 /= ws; w1 /= ws; w2 /= ws;
  _Float16* o = X0 + (size_t)idx * Kpad;
  o[0] = (_Float16)qx; o[1] = (_Float16)qy; o[2] = (_Float16)qz;
  o[3] = (_Float16)nxyz1[idx * 3 + 0];
  o[4] = (_Float16)nxyz1[idx * 3 + 1];
  o[5] = (_Float16)nxyz1[idx * 3 + 2];
  const float* p1 = pts1 + (size_t)idx * C1;
  for (int c = 0; c < C1; ++c) o[6 + c] = (_Float16)p1[c];
  const float* N2 = nxyz2 + (size_t)b * n2 * 3;
  for (int d = 0; d < 3; ++d)
    o[6 + C1 + d] = (_Float16)(w0 * N2[bi0 * 3 + d] + w1 * N2[bi1 * 3 + d] + w2 * N2[bi2 * 3 + d]);
  const float* P2 = pts2 + (size_t)b * n2 * C2;
  int base = 6 + C1 + 3;
  for (int c = 0; c < C2; ++c)
    o[base + c] = (_Float16)(w0 * P2[(size_t)bi0 * C2 + c] +
                             w1 * P2[(size_t)bi1 * C2 + c] +
                             w2 * P2[(size_t)bi2 * C2 + c]);
  for (int c = base + C2; c < Kpad; ++c) o[c] = (_Float16)0.f;
}

// ------------------------------ MLP layers ---------------------------------

__global__ void prep_weight_kernel(const float* __restrict__ w, int OC, int IC,
                                   int Kpad, _Float16* __restrict__ Wh) {
  long i = (long)blockIdx.x * blockDim.x + threadIdx.x;
  if (i >= (long)OC * Kpad) return;
  int k = (int)(i % Kpad), oc = (int)(i / Kpad);
  Wh[i] = (k < IC) ? (_Float16)w[(size_t)oc * IC + k] : (_Float16)0.f;
}

// Fragment containers (anonymous unions stay in VGPRs under full unrolling).
struct FragA { union { v16h v; u32 u[8]; }; };
struct FragB { union { v16h v; u32 u[8]; }; };

__device__ __forceinline__ void load_a(FragA& f, const u32* xr, int kb, int hi) {
  // A 16x32 f16 fragment: dwords {0..3}+4*hi then {8..11}+4*hi of the K-tile.
  #pragma unroll
  for (int r = 0; r < 4; ++r) f.u[r]     = xr[kb + r + hi * 4];
  #pragma unroll
  for (int r = 0; r < 4; ++r) f.u[4 + r] = xr[kb + 8 + r + hi * 4];
}

__device__ __forceinline__ void load_b4(FragB* fb, const u32* const* wr,
                                        int kb, int hi) {
  // B 32x16 f16 fragment: dwords {0..3}+4*hi then {8..11}+4*hi of the K-tile.
  #pragma unroll
  for (int t = 0; t < 4; ++t)
    #pragma unroll
    for (int r = 0; r < 8; ++r) fb[t].u[r] = wr[t][kb + (r & 3) + ((r >> 2) << 3) + hi * 4];
}

__device__ __forceinline__ void wmma4(v8f* c, const FragA& a, const FragB* b) {
  #pragma unroll
  for (int t = 0; t < 4; ++t)
    c[t] = __builtin_amdgcn_wmma_f32_16x16x32_f16(false, a.v, false, b[t].v,
                                                  (short)0, c[t], false, false);
}

// One wave computes a 16(M) x 64(N) strip: one shared A fragment feeds 4
// WMMAs per 32-deep K step; 2-stage pipeline overlaps loads with matrix ops.
__global__ __launch_bounds__(256)
void wmma_gemm_kernel(const _Float16* __restrict__ X, const _Float16* __restrict__ W,
                      const float* __restrict__ bias, float* __restrict__ Y,
                      int M, int Kpad, int OC) {
  int wave = (int)((blockIdx.x * blockDim.x + threadIdx.x) >> 5);
  int lane = threadIdx.x & 31;
  int mT = M >> 4, nT = OC >> 6;          // N strips of 64
  if (wave >= mT * nT) return;
  int m0 = (wave % mT) << 4;
  int n0 = (wave / mT) << 6;
  int hi = lane >> 4;
  int lm = lane & 15;
  const u32* xr = (const u32*)(X + (size_t)(m0 + lm) * Kpad);
  const u32* wr[4];
  #pragma unroll
  for (int t = 0; t < 4; ++t)
    wr[t] = (const u32*)(W + (size_t)(n0 + 16 * t + lm) * Kpad);

  v8f c[4];
  #pragma unroll
  for (int t = 0; t < 4; ++t) c[t] = (v8f){};

  int Kd = Kpad >> 1;                     // dwords per row
  int remaining = Kd >> 4;                // 32-deep K steps (>= 1)
  int kb = 0;

  FragA a0, a1;
  FragB b0[4], b1[4];
  load_a(a0, xr, kb, hi);
  load_b4(b0, wr, kb, hi);
  kb += 16;

  while (remaining >= 2) {
    __builtin_prefetch(xr + kb + 16, 0, 1);        // global_prefetch_b8
    load_a(a1, xr, kb, hi);
    load_b4(b1, wr, kb, hi);
    kb += 16;
    wmma4(c, a0, b0);                              // overlaps with a1/b1 loads
    --remaining;
    if (remaining >= 2) {
      __builtin_prefetch(xr + kb + 16, 0, 1);
      load_a(a0, xr, kb, hi);
      load_b4(b0, wr, kb, hi);
      kb += 16;
      wmma4(c, a1, b1);
      --remaining;
    } else {
      wmma4(c, a1, b1);
      --remaining;                                 // -> 0, skip tail
    }
  }
  if (remaining == 1) wmma4(c, a0, b0);

  #pragma unroll
  for (int t = 0; t < 4; ++t) {
    int col = n0 + 16 * t + lm;
    float bv = bias[col];
    #pragma unroll
    for (int r = 0; r < 8; ++r) {
      int row = m0 + r + hi * 8;
      Y[(size_t)row * OC + col] = c[t][r] + bv;
    }
  }
}

// GroupNorm stats span (all rows of a batch) x (channels of group).
__global__ void gn_stats_kernel(const float* __restrict__ Y, int OC, int Mb,
                                int cpg, int G, float* __restrict__ stats) {
  int bg = blockIdx.x;
  int b = bg / G, g = bg % G;
  int c0 = g * cpg;
  size_t row0 = (size_t)b * Mb;
  long total = (long)Mb * cpg;
  float s = 0.f, ss = 0.f;
  for (long i = (long)blockIdx.y * blockDim.x + threadIdx.x; i < total;
       i += (long)gridDim.y * blockDim.x) {
    long r = i / cpg;
    int  c = c0 + (int)(i % cpg);
    float v = Y[(row0 + r) * OC + c];
    s += v; ss += v * v;
  }
  __shared__ float sh[256], sh2[256];
  sh[threadIdx.x] = s; sh2[threadIdx.x] = ss;
  __syncthreads();
  for (int st = 128; st > 0; st >>= 1) {
    if ((int)threadIdx.x < st) {
      sh [threadIdx.x] += sh [threadIdx.x + st];
      sh2[threadIdx.x] += sh2[threadIdx.x + st];
    }
    __syncthreads();
  }
  if (threadIdx.x == 0) {
    atomicAdd(&stats[bg * 2 + 0], sh[0]);
    atomicAdd(&stats[bg * 2 + 1], sh2[0]);
  }
}

__global__ void gn_apply_relu_kernel(float* __restrict__ Y,
                                     const float* __restrict__ stats,
                                     const float* __restrict__ gamma,
                                     const float* __restrict__ beta,
                                     long total, int OC, int Mb, int cpg, int G) {
  long i = (long)blockIdx.x * blockDim.x + threadIdx.x;
  if (i >= total) return;
  int c = (int)(i % OC);
  long r = i / OC;
  int b = (int)(r / Mb);
  int bg = b * G + c / cpg;
  float cnt = (float)Mb * (float)cpg;
  float m = stats[bg * 2 + 0] / cnt;
  float v = stats[bg * 2 + 1] / cnt - m * m;
  float x = (Y[i] - m) * rsqrtf(v + 1e-5f) * gamma[c] + beta[c];
  Y[i] = fmaxf(x, 0.f);
}

__global__ void cvt_f32_to_f16_kernel(const float* __restrict__ s,
                                      _Float16* __restrict__ d, long n) {
  long i = (long)blockIdx.x * blockDim.x + threadIdx.x;
  if (i < n) d[i] = (_Float16)s[i];
}

__global__ void maxpool_kernel(const float* __restrict__ Y, int ns, int OC,
                               float* __restrict__ out, long total) {
  long i = (long)blockIdx.x * blockDim.x + threadIdx.x;   // srow*OC + c
  if (i >= total) return;
  int  c = (int)(i % OC);
  long s = i / OC;
  const float* base = Y + (s * ns) * (size_t)OC + c;
  float m = -1e30f;
  for (int k = 0; k < ns; ++k) m = fmaxf(m, base[(size_t)k * OC]);
  out[i] = m;
}

// Final 64->6 conv + tanh, split into depth_delta (scaled) and unit normal.
__global__ void final_kernel(const float* __restrict__ p2,
                             const float* __restrict__ cw,
                             const float* __restrict__ cb,
                             float* __restrict__ out, int total) {
  int i = blockIdx.x * blockDim.x + threadIdx.x;   // b*N + j
  if (i >= total) return;
  const float* p = p2 + (size_t)i * 64;
  float o[6];
  for (int oc = 0; oc < 6; ++oc) {
    float s = cb[oc];
    const float* w = cw + oc * 64;
    for (int c = 0; c < 64; ++c) s += p[c] * w[c];
    o[oc] = tanhf(s);
  }
  out[(size_t)i * 3 + 0] = o[0] * 0.25f;
  out[(size_t)i * 3 + 1] = o[1] * 0.25f;
  out[(size_t)i * 3 + 2] = o[2] * 0.25f;
  float nx = o[3], ny = o[4], nz = o[5];
  float nn = fmaxf(sqrtf(nx * nx + ny * ny + nz * nz), 1e-10f);
  float* nrm = out + (size_t)total * 3;
  nrm[(size_t)i * 3 + 0] = nx / nn;
  nrm[(size_t)i * 3 + 1] = ny / nn;
  nrm[(size_t)i * 3 + 2] = nz / nn;
}

// ------------------------------ host driver --------------------------------

static inline int cdiv(long a, long b) { return (int)((a + b - 1) / b); }
static inline int al32(int x) { return (x + 31) & ~31; }

extern "C" void kernel_launch(void* const* d_in, const int* in_sizes, int n_in,
                              void* d_out, int out_size, void* d_ws, size_t ws_size,
                              hipStream_t stream) {
  const float* in_xyz  = (const float*)d_in[0];
  const float* in_nxyz = (const float*)d_in[1];
  const float* in_pts  = (const float*)d_in[2];
  // params flattened in insertion order: 15 layers x {w,b,g,be}, then conv_w, conv_b
  auto P = [&](int i) { return (const float*)d_in[i]; };
  auto LW = [&](int L) { return 3 + 4 * L; };

  // ---- workspace carve (bump allocator, 256B aligned) ----
  char* base = (char*)d_ws;
  size_t off = 0;
  auto carve = [&](size_t bytes) -> void* {
    void* p = base + off;
    off += (bytes + 255) & ~(size_t)255;
    return p;
  };
  const int nl[5] = {8192, 1024, 256, 64, 16};
  const int cl[5] = {6, 64, 128, 256, 512};
  float *xyzL[5], *nxyzL[5], *featL[5];
  for (int l = 0; l < 5; ++l) {
    xyzL[l]  = (float*)carve((size_t)BATCH * nl[l] * 3 * 4);
    nxyzL[l] = (float*)carve((size_t)BATCH * nl[l] * 3 * 4);
    featL[l] = (float*)carve((size_t)BATCH * nl[l] * cl[l] * 4);
  }
  float* dist = (float*)carve((size_t)BATCH * NPTS * 4);
  int*   fidx = (int*)  carve((size_t)BATCH * 1024 * 4);
  int*   gidx = (int*)  carve((size_t)BATCH * 1024 * 64 * 4);
  _Float16* X0 = (_Float16*)carve((size_t)8388608  * 2);   // max rows*Kpad (f16)
  _Float16* X1 = (_Float16*)carve((size_t)16777216 * 2);   // inter-layer acts
  float* Y     = (float*)   carve((size_t)16777216 * 4);   // GEMM output
  _Float16* Wh = (_Float16*)carve((size_t)512 * 800 * 2);  // padded f16 weights
  float* stats = (float*)   carve((size_t)BATCH * 32 * 2 * 4);
  float* p2a   = (float*)   carve((size_t)BATCH * NPTS * 64 * 4);
  float* p2b   = (float*)   carve((size_t)BATCH * NPTS * 64 * 4);
  (void)ws_size; (void)n_in; (void)in_sizes; (void)out_size;

  // One MLP layer: f16 panel -> WMMA GEMM -> GroupNorm stats -> norm+ReLU.
  auto run_layer = [&](const _Float16* Xp, int Kpad, int IC, int M, int OC,
                       int G, int L) {
    prep_weight_kernel<<<cdiv((long)OC * Kpad, 256), 256, 0, stream>>>(
        P(LW(L)), OC, IC, Kpad, Wh);
    int tiles = (M / 16) * (OC / 64);
    wmma_gemm_kernel<<<cdiv(tiles, 8), 256, 0, stream>>>(
        Xp, Wh, P(LW(L) + 1), Y, M, Kpad, OC);
    hipMemsetAsync(stats, 0, (size_t)BATCH * G * 2 * sizeof(float), stream);
    int Mb = M / BATCH, cpg = OC / G;
    dim3 sg(BATCH * G, 16);
    gn_stats_kernel<<<sg, 256, 0, stream>>>(Y, OC, Mb, cpg, G, stats);
    long tot = (long)M * OC;
    gn_apply_relu_kernel<<<cdiv(tot, 256), 256, 0, stream>>>(
        Y, stats, P(LW(L) + 2), P(LW(L) + 3), tot, OC, Mb, cpg, G);
  };

  // ---- stage 0: transpose (B,C,N) -> (B,N,C) level-0 features ----
  transpose_in_kernel<<<cdiv(BATCH * NPTS, 256), 256, 0, stream>>>(
      in_xyz, in_nxyz, in_pts, xyzL[0], nxyzL[0], featL[0]);

  // ---- set abstraction levels ----
  const int   npA[4]  = {1024, 256, 64, 16};
  const int   nsA[4]  = {64, 32, 32, 32};
  const float r2A[4]  = {0.01f, 0.04f, 0.16f, 0.64f};
  const int   icA[4]  = {12, 70, 134, 262};
  const int   ocA[4][2] = {{64, 64}, {128, 128}, {256, 256}, {512, 512}};
  const int   gA[4][2]  = {{4, 4}, {8, 8}, {16, 16}, {32, 32}};
  int L = 0;
  for (int l = 0; l < 4; ++l) {
    int n = nl[l], np = npA[l], ns = nsA[l];
    fps_kernel<<<BATCH, 256, 0, stream>>>(xyzL[l], n, np, fidx, dist);
    gather_pts_kernel<<<cdiv(BATCH * np, 256), 256, 0, stream>>>(
        xyzL[l], nxyzL[l], fidx, n, np, xyzL[l + 1], nxyzL[l + 1], BATCH * np);
    ball_query_kernel<<<cdiv(BATCH * np, 128), 128, 0, stream>>>(
        xyzL[l], xyzL[l + 1], n, np, ns, r2A[l], gidx, BATCH * np);
    int M = BATCH * np * ns;
    int Kp = al32(icA[l]);
    build_sa_input_kernel<<<cdiv(M, 256), 256, 0, stream>>>(
        xyzL[l], nxyzL[l], featL[l], cl[l], xyzL[l + 1], gidx, n, np, ns, Kp, X0, M);
    run_layer(X0, Kp, icA[l], M, ocA[l][0], gA[l][0], L); ++L;
    cvt_f32_to_f16_kernel<<<cdiv((long)M * ocA[l][0], 256), 256, 0, stream>>>(
        Y, X1, (long)M * ocA[l][0]);
    run_layer(X1, ocA[l][0], ocA[l][0], M, ocA[l][1], gA[l][1], L); ++L;
    maxpool_kernel<<<cdiv((long)BATCH * np * ocA[l][1], 256), 256, 0, stream>>>(
        Y, ns, ocA[l][1], featL[l + 1], (long)BATCH * np * ocA[l][1]);
  }

  // ---- feature propagation levels ----
  const int icF[4]    = {777, 393, 201, 79};
  const int ocF[4][2] = {{512, 256}, {256, 128}, {128, 64}, {64, -1}};
  const int gF[4][2]  = {{32, 16}, {16, 8}, {8, 4}, {4, -1}};
  float* p2 = featL[4];
  int C2 = 512;
  float* pa = p2a; float* pb = p2b;
  for (int lvl = 0; lvl < 4; ++lvl) {
    int l1 = 3 - lvl, l2 = 4 - lvl;
    int n1 = nl[l1], n2 = nl[l2], C1 = cl[l1];
    int Kp = al32(icF[lvl]);
    int M = BATCH * n1;
    fp_build_kernel<<<cdiv(M, 128), 128, 0, stream>>>(
        xyzL[l1], nxyzL[l1], featL[l1], C1, xyzL[l2], nxyzL[l2], p2, C2,
        n1, n2, Kp, X0, M);
    run_layer(X0, Kp, icF[lvl], M, ocF[lvl][0], gF[lvl][0], L); ++L;
    int Cout;
    if (ocF[lvl][1] > 0) {
      cvt_f32_to_f16_kernel<<<cdiv((long)M * ocF[lvl][0], 256), 256, 0, stream>>>(
          Y, X1, (long)M * ocF[lvl][0]);
      run_layer(X1, ocF[lvl][0], ocF[lvl][0], M, ocF[lvl][1], gF[lvl][1], L); ++L;
      Cout = ocF[lvl][1];
    } else {
      Cout = ocF[lvl][0];
    }
    hipMemcpyAsync(pa, Y, (size_t)M * Cout * sizeof(float),
                   hipMemcpyDeviceToDevice, stream);
    p2 = pa; C2 = Cout;
    float* tmp = pa; pa = pb; pb = tmp;
  }

  // ---- head: tanh(conv 64->6), depth_delta + unit normals ----
  final_kernel<<<cdiv(BATCH * NPTS, 128), 128, 0, stream>>>(
      p2, P(3 + 15 * 4), P(3 + 15 * 4 + 1), (float*)d_out, BATCH * NPTS);
}
